// MelDecoder_68650757259803
// MI455X (gfx1250) — compile-verified, matrix operand order checked by reading, not run on hardware
//
#include <hip/hip_runtime.h>

typedef __attribute__((ext_vector_type(16))) __bf16 v16bf;
typedef __attribute__((ext_vector_type(8)))  float  v8f;
typedef unsigned short u16;
typedef unsigned int   u32;

#define BB    32
#define TE    512
#define TM    600
#define ENCD  512
#define DECD  1024
#define NMEL  80
#define PRE   256
#define ATTD  128
#define LOCD  32
#define LOCK  31
#define PDIM  512
#define K1    1792   // 256 (prenet) + 512 (ctx) + 1024 (h1)
#define K2    2048   // 1024 (h1) + 1024 (h2)
#define G4    4096
#define PLDT  616    // 600 + conv pad + slack (t-major time stride)

// ---------- bf16 helpers (bit-exact, no reliance on __bf16 arithmetic) ----------
__device__ __forceinline__ float bf2f(u16 u){ union{u32 i; float f;} z; z.i = ((u32)u) << 16; return z.f; }
__device__ __forceinline__ u16 f2bf(float f){
  union{float f; u32 i;} z; z.f = f; u32 u = z.i;
  return (u16)((u + 0x7FFFu + ((u >> 16) & 1u)) >> 16);
}
__device__ __forceinline__ __bf16 usbf(u16 u){ union{u16 s; __bf16 b;} z; z.s = u; return z.b; }

// ISA 7.12.2: 16-bit A 16x32 — lane(0..15)=row M, half=lane>>4 selects K in
// {0..7,16..23} (half 0) / {8..15,24..31} (half 1), packed 2 per VGPR.
__device__ __forceinline__ int kpat(int v, int half){ return ((v < 4) ? 2*v : 16 + 2*(v-4)) + half*8; }

// Gather one 16x32 bf16 fragment. rstride: stride between M (or N) rows,
// kstride: stride between K elements. With kstride==1 the 16 elements are two
// contiguous 16B runs per lane -> b128 loads.
__device__ __forceinline__ v16bf frag_gather(const u16* p0, long kstride, long rstride){
  int lane = threadIdx.x & 31;
  int r = lane & 15, half = lane >> 4;
  const u16* p = p0 + (long)r * rstride;
  v16bf f;
#pragma unroll
  for (int v = 0; v < 8; ++v){
    int k = kpat(v, half);
    f[2*v]   = usbf(p[(long)k * kstride]);
    f[2*v+1] = usbf(p[(long)(k+1) * kstride]);
  }
  return f;
}

__device__ __forceinline__ v8f wmma_bf16(v16bf a, v16bf b, v8f c){
  return __builtin_amdgcn_wmma_f32_16x16x32_bf16(false, a, false, b, (short)0, c, false, false);
}

// ---------- one-time prep kernels ----------
__global__ void kzero_u32(u32* p, long n){
  long i = (long)blockIdx.x * blockDim.x + threadIdx.x;
  long st = (long)gridDim.x * blockDim.x;
  for (; i < n; i += st) p[i] = 0u;
}

__global__ void kconv_bf16(const float* s, u16* d, long n){
  long i = (long)blockIdx.x * blockDim.x + threadIdx.x;
  long st = (long)gridDim.x * blockDim.x;
  for (; i < n; i += st) d[i] = f2bf(s[i]);
}

// concat [wih | whh] along K into one [4096][kx+kh] bf16 matrix
__global__ void kbuild_wcat(const float* wih, const float* whh, u16* d, int kx, int kh){
  int K = kx + kh;
  long n = (long)G4 * K;
  long i = (long)blockIdx.x * blockDim.x + threadIdx.x;
  long st = (long)gridDim.x * blockDim.x;
  for (; i < n; i += st){
    long row = i / K; int c = (int)(i - row * K);
    float v = (c < kx) ? wih[row * kx + c] : whh[row * kh + (c - kx)];
    d[i] = f2bf(v);
  }
}

__global__ void kbias(const float* a, const float* b, float* d, int n){
  int i = blockIdx.x * blockDim.x + threadIdx.x;
  if (i < n) d[i] = a[i] + b[i];
}

// postnet weights [cout][cin][5] -> [cout][5][cinP] bf16 (ci contiguous, zero-pad)
__global__ void kconv_pnw(const float* s, u16* d, int cin, int cinP, int cout){
  long n = (long)cout * 5 * cinP;
  long i = (long)blockIdx.x * blockDim.x + threadIdx.x;
  long st = (long)gridDim.x * blockDim.x;
  for (; i < n; i += st){
    int co = (int)(i / (5 * cinP)); int rem = (int)(i % (5 * cinP));
    int k = rem / cinP, ci = rem % cinP;
    d[i] = f2bf(ci < cin ? s[((long)co * cin + ci) * 5 + k] : 0.f);
  }
}

// proc_keys = enc @ k_w^T   (fp32 VALU, one-time)
__global__ void __launch_bounds__(128) kpk(const float* enc, const float* kw, float* pk){
  int bt = blockIdx.x, a = threadIdx.x;
  const float* e = enc + (long)bt * ENCD;
  const float* w = kw + (long)a * ENCD;
  float acc = 0.f;
  for (int d = 0; d < ENCD; ++d) acc += e[d] * w[d];
  pk[(long)bt * ATTD + a] = acc;
}

// ---------- per-step: prenet + location-sensitive attention (1 block / batch) ----------
__global__ void __launch_bounds__(256) katt(
    const float* mels, int t, const float* enc, const int* elen,
    const float* w1, const float* b1, const float* w2, const float* b2,
    const float* qw, const float* lconv, const u16* locw, const float* vw,
    const float* pk, u16* X1p, const u16* X2p, float* attnw, float* ctx,
    float* alignsOut)
{
  __shared__ float s_attn[TE + LOCK - 1];
  __shared__ u16   s_loc[TE * LOCD];
  __shared__ float s_e[TE];
  __shared__ float s_part[TE * 8];   // per-(t, n-tile) deterministic partials
  __shared__ float s_q[ATTD];
  __shared__ float s_p1[PRE];
  __shared__ float s_red[256];
  int b = blockIdx.x, tid = threadIdx.x;

  // prenet layer1 (K=80)
  {
    const float* wr = w1 + tid * NMEL;
    float acc = b1[tid];
    for (int k = 0; k < NMEL; ++k) acc += mels[((long)b*NMEL + k)*TM + t] * wr[k];
    s_p1[tid] = fmaxf(acc, 0.f);
  }
  // query projection from previous h2 (bf16 in X2p[:,1024:2048))
  if (tid < ATTD){
    const float* wr = qw + (long)tid * DECD;
    const u16* h2 = X2p + (long)b * K2 + DECD;
    float acc = 0.f;
    for (int k = 0; k < DECD; ++k) acc += bf2f(h2[k]) * wr[k];
    s_q[tid] = acc;
  }
  for (int i = tid; i < TE + LOCK - 1; i += 256){
    int tt = i - (LOCK - 1) / 2;
    s_attn[i] = (tt >= 0 && tt < TE) ? attnw[b*TE + tt] : 0.f;
  }
  __syncthreads();

  // prenet layer2 -> X1[:,0:256) bf16
  {
    const float* wr = w2 + tid * PRE;
    float acc = b2[tid];
    for (int k = 0; k < PRE; ++k) acc += s_p1[k] * wr[k];
    X1p[(long)b*K1 + tid] = f2bf(fmaxf(acc, 0.f));
  }
  // location conv (1 -> 32 channels, K=31), store t-major bf16 for WMMA A
  for (int i = tid; i < TE * LOCD; i += 256){
    int tt = i >> 5, c = i & 31;
    const float* wk = lconv + c * LOCK;
    float acc = 0.f;
    for (int k = 0; k < LOCK; ++k) acc += s_attn[tt + k] * wk[k];
    s_loc[tt * LOCD + c] = f2bf(acc);
  }
  __syncthreads();

  // WMMA: locp[512,128] = s_loc[512,32] @ loc_w^T ; fuse tanh-energy reduce.
  // Deterministic: per tile row, sum 16 lane contributions by shfl_xor
  // (masks 1..8 stay inside the 16-lane half-group on wave32).
  int wave = tid >> 5, lane = tid & 31;
  const float* pkb = pk + (long)b * TE * ATTD;
  for (int task = wave; task < 32 * 8; task += 8){
    int tm = task >> 3, tn = task & 7;
    v16bf af = frag_gather(s_loc + tm*16*LOCD, 1, LOCD);
    v16bf bf = frag_gather(locw + tn*16*LOCD, 1, LOCD);
    v8f dacc = {};
    dacc = wmma_bf16(af, bf, dacc);
    int n = lane & 15; int mb = (lane >> 4) * 8;
    int aa = tn*16 + n;
    float qa = s_q[aa], va = vw[aa];
#pragma unroll
    for (int v = 0; v < 8; ++v){
      int ttg = tm*16 + mb + v;
      float cv = va * tanhf(dacc[v] + qa + pkb[(long)ttg*ATTD + aa]);
      cv += __shfl_xor(cv, 1, 32);
      cv += __shfl_xor(cv, 2, 32);
      cv += __shfl_xor(cv, 4, 32);
      cv += __shfl_xor(cv, 8, 32);
      if ((lane & 15) == 0) s_part[ttg * 8 + tn] = cv;
    }
  }
  __syncthreads();

  // masked softmax over TE (fixed-order partial sum -> deterministic)
  int len = elen[b];
  float mx = -1e30f;
  for (int i = tid; i < TE; i += 256){
    const float* pp = s_part + i * 8;
    float e = ((pp[0] + pp[1]) + (pp[2] + pp[3])) + ((pp[4] + pp[5]) + (pp[6] + pp[7]));
    s_e[i] = e;
    if (i < len) mx = fmaxf(mx, e);
  }
  s_red[tid] = mx; __syncthreads();
  for (int s = 128; s > 0; s >>= 1){ if (tid < s) s_red[tid] = fmaxf(s_red[tid], s_red[tid+s]); __syncthreads(); }
  mx = s_red[0]; __syncthreads();
  float sm = 0.f;
  for (int i = tid; i < TE; i += 256){
    float e = (i < len) ? __expf(s_e[i] - mx) : 0.f;
    s_e[i] = e; sm += e;
  }
  s_red[tid] = sm; __syncthreads();
  for (int s = 128; s > 0; s >>= 1){ if (tid < s) s_red[tid] += s_red[tid+s]; __syncthreads(); }
  float inv = 1.f / s_red[0];
  __syncthreads();
  for (int i = tid; i < TE; i += 256){
    float a = s_e[i] * inv;
    s_e[i] = a;
    attnw[b*TE + i] = a;
    alignsOut[((long)b*TM + t)*TE + i] = a;
  }
  __syncthreads();

  // context = attn @ enc[b]   -> ctx fp32 + X1[:,256:768) bf16
  const float* eb = enc + (long)b * TE * ENCD;
  for (int d0 = tid; d0 < ENCD; d0 += 256){
    float acc = 0.f;
    for (int tt = 0; tt < TE; ++tt) acc += s_e[tt] * eb[(long)tt*ENCD + d0];
    ctx[b*ENCD + d0] = acc;
    X1p[(long)b*K1 + PRE + d0] = f2bf(acc);
  }
}

// ---------- per-step: fused LSTM gates (WMMA) + cell update ----------
// Block bl owns gate-aligned cols {bl*32..+31} at offsets {0,1024,2048,3072},
// so the cell nonlinearity is computed from LDS in the same kernel.
// X is staged into LDS once per block; only L2-resident bf16 weights stream.
__global__ void __launch_bounds__(256) kgates(
    const u16* X, int ldx, int nk, const u16* W, const float* bias,
    float* c, u16* hA, int ldA, int offA, u16* hB, int ldB, int offB)
{
  __shared__ u16   s_X[BB * K2];       // max 128KB (K2); K1 uses a prefix
  __shared__ float s_G[4][32][33];
  int bl = blockIdx.x, tid = threadIdx.x;
  int jb = bl * 32;
  int wave = tid >> 5, lane = tid & 31;

  // stage X (bf16, [32][nk]) into LDS
  {
    const u32* xs = (const u32*)X;
    u32* xd = (u32*)s_X;
    int nd = (BB * nk) >> 1;
    for (int i = tid; i < nd; i += 256) xd[i] = xs[i];
  }
  __syncthreads();

  for (int tt = wave; tt < 16; tt += 8){
    int m0   = (tt & 1) * 16;
    int nsub = (tt >> 1) & 1;
    int q    = tt >> 2;
    int n0   = q * 1024 + jb + nsub * 16;
    v8f acc = {};
    const u16* ap = s_X + (long)m0 * ldx;
    const u16* bp = W + (long)n0 * nk;
    for (int kk = 0; kk < nk; kk += 32){
      __builtin_prefetch(bp + kk + 256, 0, 1);
      v16bf af = frag_gather(ap + kk, 1, ldx);
      v16bf bf = frag_gather(bp + kk, 1, nk);
      acc = wmma_bf16(af, bf, acc);
    }
    int n = lane & 15, mb = (lane >> 4) * 8;
#pragma unroll
    for (int v = 0; v < 8; ++v)
      s_G[q][m0 + mb + v][nsub*16 + n] = acc[v];
  }
  __syncthreads();

#pragma unroll
  for (int i = 0; i < 4; ++i){
    int e = tid + i * 256;
    int bidx = e & 31, jl = e >> 5;
    int j = jb + jl;
    float gi = s_G[0][bidx][jl] + bias[j];
    float gf = s_G[1][bidx][jl] + bias[1024 + j];
    float gg = s_G[2][bidx][jl] + bias[2048 + j];
    float go = s_G[3][bidx][jl] + bias[3072 + j];
    float si = 1.f / (1.f + __expf(-gi));
    float sf = 1.f / (1.f + __expf(-gf));
    float so = 1.f / (1.f + __expf(-go));
    float cn = sf * c[bidx*DECD + j] + si * tanhf(gg);
    c[bidx*DECD + j] = cn;
    u16 hb = f2bf(so * tanhf(cn));
    hA[(long)bidx*ldA + offA + j] = hb;
    if (hB) hB[(long)bidx*ldB + offB + j] = hb;
  }
}

// ---------- per-step: mel + stop projection ----------
__global__ void __launch_bounds__(128) kout(
    const u16* X2n, const float* ctx, const float* melw, const float* melb,
    const float* stopw, const float* stopb, float* outMel, float* outStop, int t)
{
  int b = blockIdx.x, tid = threadIdx.x;
  if (tid > 80) return;
  const float* wr = (tid < 80) ? (melw + (long)tid * (DECD + ENCD)) : stopw;
  const u16* h2 = X2n + (long)b * K2 + DECD;
  const float* cb = ctx + b * ENCD;
  float acc = (tid < 80) ? melb[tid] : stopb[0];
  for (int k = 0; k < DECD; ++k) acc += bf2f(h2[k]) * wr[k];
  for (int k = 0; k < ENCD; ++k) acc += cb[k] * wr[DECD + k];
  if (tid < 80) outMel[((long)b*NMEL + tid)*TM + t] = acc;
  else          outStop[b*TM + t] = acc;
}

// ---------- postnet (activations t-major: [b][PLDT][C], channels contiguous) ----------
__global__ void kpin0(const float* mel, u16* d){
  long n = (long)BB * NMEL * TM;
  long i = (long)blockIdx.x * blockDim.x + threadIdx.x;
  long st = (long)gridDim.x * blockDim.x;
  for (; i < n; i += st){
    int t = (int)(i % TM); long r = i / TM;
    int ci = (int)(r % NMEL); int b = (int)(r / NMEL);
    d[((long)b*PLDT + 2 + t)*96 + ci] = f2bf(mel[i]);
  }
}

// conv1d(K=5,'same') as a 5-tap WMMA GEMM over channels; epilogue does
// bias + BN affine (+tanh -> next padded bf16 buffer, or +residual -> mel_post).
__global__ void __launch_bounds__(256) kpconv(
    const u16* xin, int cinP, const u16* w, const float* bias, const float* gamma,
    const float* beta, int cout, u16* yout, float* ypost, const float* melref, int ntasks)
{
  int wave = threadIdx.x >> 5, lane = threadIdx.x & 31;
  int task = blockIdx.x * 8 + wave;
  if (task >= ntasks) return;
  int ntiles = cout >> 4;
  int nt = task % ntiles; int r = task / ntiles;
  int mt = r % 38; int b = r / 38;
  int t0 = mt * 16, co0 = nt * 16;
  v8f acc = {};
  for (int k = 0; k < 5; ++k){
    const u16* ap = xin + ((long)b * PLDT + t0 + k) * cinP;   // A row m = t
    const u16* bp = w + ((long)co0 * 5 + k) * cinP;           // [co][k][ci]
    for (int ci = 0; ci < cinP; ci += 32){
      v16bf af = frag_gather(ap + ci, 1, cinP);
      v16bf bf = frag_gather(bp + ci, 1, (long)5 * cinP);
      acc = wmma_bf16(af, bf, acc);
    }
  }
  int n = lane & 15, mb = (lane >> 4) * 8;
  int co = co0 + n;
  float sc = gamma[co] * rsqrtf(1.f + 1e-5f);
  float sh = beta[co], bi = bias[co];
#pragma unroll
  for (int v = 0; v < 8; ++v){
    int t = t0 + mb + v;
    if (t >= TM) continue;
    float val = (acc[v] + bi) * sc + sh;
    if (yout){
      yout[((long)b*PLDT + 2 + t)*PDIM + co] = f2bf(tanhf(val));
    } else {
      long idx = ((long)b*NMEL + co)*TM + t;
      ypost[idx] = melref[idx] + val;
    }
  }
}

// =======================================================================
extern "C" void kernel_launch(void* const* d_in, const int* in_sizes, int n_in,
                              void* d_out, int out_size, void* d_ws, size_t ws_size,
                              hipStream_t stream)
{
  (void)in_sizes; (void)n_in; (void)out_size; (void)ws_size;
  const float* enc   = (const float*)d_in[0];
  const int*   elen  = (const int*)  d_in[1];
  const float* mels  = (const float*)d_in[2];
  const float* pw1   = (const float*)d_in[3];
  const float* pb1   = (const float*)d_in[4];
  const float* pw2   = (const float*)d_in[5];
  const float* pb2   = (const float*)d_in[6];
  const float* qw    = (const float*)d_in[7];
  const float* kw    = (const float*)d_in[8];
  const float* lconv = (const float*)d_in[9];
  const float* locw  = (const float*)d_in[10];
  const float* vw    = (const float*)d_in[11];
  const float* wih1  = (const float*)d_in[12];
  const float* whh1  = (const float*)d_in[13];
  const float* bih1  = (const float*)d_in[14];
  const float* bhh1  = (const float*)d_in[15];
  const float* wih2  = (const float*)d_in[16];
  const float* whh2  = (const float*)d_in[17];
  const float* bih2  = (const float*)d_in[18];
  const float* bhh2  = (const float*)d_in[19];
  const float* melw  = (const float*)d_in[20];
  const float* melb  = (const float*)d_in[21];
  const float* stw   = (const float*)d_in[22];
  const float* stb   = (const float*)d_in[23];
  const float *pnw[5], *pnb[5], *png[5], *pnbt[5];
  for (int i = 0; i < 5; ++i){
    pnw[i]  = (const float*)d_in[24 + 4*i];
    pnb[i]  = (const float*)d_in[25 + 4*i];
    png[i]  = (const float*)d_in[26 + 4*i];
    pnbt[i] = (const float*)d_in[27 + 4*i];
  }

  float* out      = (float*)d_out;
  float* outMel   = out;                 // [32,80,600]
  float* outPost  = out + 1536000;       // [32,80,600]
  float* outStop  = out + 3072000;       // [32,600]
  float* outAlign = out + 3091200;       // [32,600,512]

  char* ws = (char*)d_ws;
  size_t cur = 0;
  auto alloc = [&](size_t bytes)->char* {
    char* p = ws + cur;
    cur = (cur + bytes + 255) & ~(size_t)255;
    return p;
  };
  u16*   wcat1  = (u16*)  alloc((size_t)G4 * K1 * 2);
  u16*   wcat2  = (u16*)  alloc((size_t)G4 * K2 * 2);
  float* bias1  = (float*)alloc(4096 * 4);
  float* bias2  = (float*)alloc(4096 * 4);
  u16*   locwb  = (u16*)  alloc(128 * 32 * 2);
  float* pkbuf  = (float*)alloc((size_t)BB * TE * ATTD * 4);
  u16*   pnwb0  = (u16*)  alloc((size_t)512 * 5 * 96 * 2);
  u16*   pnwb1  = (u16*)  alloc((size_t)512 * 5 * 512 * 2);
  u16*   pnwb2  = (u16*)  alloc((size_t)512 * 5 * 512 * 2);
  u16*   pnwb3  = (u16*)  alloc((size_t)512 * 5 * 512 * 2);
  u16*   pnwb4  = (u16*)  alloc((size_t)80 * 5 * 512 * 2);
  char*  stateS = ws + cur;
  u16*   X1buf[2]; u16* X2buf[2];
  X1buf[0] = (u16*)alloc((size_t)BB * K1 * 2);
  X1buf[1] = (u16*)alloc((size_t)BB * K1 * 2);
  X2buf[0] = (u16*)alloc((size_t)BB * K2 * 2);
  X2buf[1] = (u16*)alloc((size_t)BB * K2 * 2);
  float* c1     = (float*)alloc(BB * DECD * 4);
  float* c2     = (float*)alloc(BB * DECD * 4);
  float* attnwb = (float*)alloc(BB * TE * 4);
  float* ctxb   = (float*)alloc(BB * ENCD * 4);
  char*  stateE = ws + cur;
  char*  pbS    = ws + cur;
  u16*   pbuf0  = (u16*)alloc((size_t)BB * PLDT * 96 * 2);
  u16*   pbufA  = (u16*)alloc((size_t)BB * PLDT * 512 * 2);
  u16*   pbufB  = (u16*)alloc((size_t)BB * PLDT * 512 * 2);
  char*  pbE    = ws + cur;

  // ---- one-time prep ----
  {
    long n1 = (long)G4 * K1, n2 = (long)G4 * K2;
    kbuild_wcat<<<(int)((n1 + 255) / 256), 256, 0, stream>>>(wih1, whh1, wcat1, 768, 1024);
    kbuild_wcat<<<(int)((n2 + 255) / 256), 256, 0, stream>>>(wih2, whh2, wcat2, 1024, 1024);
    kbias<<<16, 256, 0, stream>>>(bih1, bhh1, bias1, 4096);
    kbias<<<16, 256, 0, stream>>>(bih2, bhh2, bias2, 4096);
    kconv_bf16<<<16, 256, 0, stream>>>(locw, locwb, 128 * 32);
    kpk<<<BB * TE, 128, 0, stream>>>(enc, kw, pkbuf);
    kconv_pnw<<<960, 256, 0, stream>>>(pnw[0], pnwb0, 80, 96, 512);
    kconv_pnw<<<5120, 256, 0, stream>>>(pnw[1], pnwb1, 512, 512, 512);
    kconv_pnw<<<5120, 256, 0, stream>>>(pnw[2], pnwb2, 512, 512, 512);
    kconv_pnw<<<5120, 256, 0, stream>>>(pnw[3], pnwb3, 512, 512, 512);
    kconv_pnw<<<800, 256, 0, stream>>>(pnw[4], pnwb4, 512, 512, 80);
    long zs = (long)(stateE - stateS) / 4;
    kzero_u32<<<(int)((zs + 255) / 256), 256, 0, stream>>>((u32*)stateS, zs);
    long zp = (long)(pbE - pbS) / 4;
    kzero_u32<<<4096, 256, 0, stream>>>((u32*)pbS, zp);
  }

  // ---- sequential decode, double-buffered activations by step parity ----
  for (int t = 0; t < TM; ++t){
    int p = t & 1;
    u16* X1p = X1buf[p];     u16* X1n = X1buf[1 - p];
    u16* X2p = X2buf[p];     u16* X2n = X2buf[1 - p];
    katt<<<BB, 256, 0, stream>>>(mels, t, enc, elen, pw1, pb1, pw2, pb2, qw,
                                 lconv, locwb, vw, pkbuf, X1p, X2p, attnwb,
                                 ctxb, outAlign);
    // layer1: h1 -> X2p[:,0:1024) (this step) and X1n[:,768:1792) (next step)
    kgates<<<32, 256, 0, stream>>>(X1p, K1, K1, wcat1, bias1, c1,
                                   X2p, K2, 0, X1n, K1, 768);
    // layer2: h2 -> X2n[:,1024:2048) (read by katt/kgates2 next step)
    kgates<<<32, 256, 0, stream>>>(X2p, K2, K2, wcat2, bias2, c2,
                                   X2n, K2, 1024, (u16*)nullptr, 0, 0);
    kout<<<BB, 128, 0, stream>>>(X2n, ctxb, melw, melb, stw, stb,
                                 outMel, outStop, t);
  }

  // ---- postnet: 5x conv(K=5) as WMMA GEMM ----
  kpin0<<<6000, 256, 0, stream>>>(outMel, pbuf0);
  int nt512 = BB * 38 * (512 / 16);
  int nt80  = BB * 38 * (80 / 16);
  kpconv<<<(nt512 + 7) / 8, 256, 0, stream>>>(pbuf0, 96,  pnwb0, pnb[0], png[0], pnbt[0], 512, pbufA, (float*)nullptr, (const float*)nullptr, nt512);
  kpconv<<<(nt512 + 7) / 8, 256, 0, stream>>>(pbufA, 512, pnwb1, pnb[1], png[1], pnbt[1], 512, pbufB, (float*)nullptr, (const float*)nullptr, nt512);
  kpconv<<<(nt512 + 7) / 8, 256, 0, stream>>>(pbufB, 512, pnwb2, pnb[2], png[2], pnbt[2], 512, pbufA, (float*)nullptr, (const float*)nullptr, nt512);
  kpconv<<<(nt512 + 7) / 8, 256, 0, stream>>>(pbufA, 512, pnwb3, pnb[3], png[3], pnbt[3], 512, pbufB, (float*)nullptr, (const float*)nullptr, nt512);
  kpconv<<<(nt80  + 7) / 8, 256, 0, stream>>>(pbufB, 512, pnwb4, pnb[4], png[4], pnbt[4], 80, (u16*)nullptr, outPost, outMel, nt80);
}
